// MultiHeadFlexAttention_19756849562256
// MI455X (gfx1250) — matrix operand order repaired and checked by reading, not compiled, not run on hardware
//
#include <hip/hip_runtime.h>
#include <hip/hip_bf16.h>

typedef _Float16 v8h  __attribute__((ext_vector_type(8)));
typedef _Float16 v16h __attribute__((ext_vector_type(16)));
typedef float    v8f  __attribute__((ext_vector_type(8)));

#define N_EMBD 1024
#define N_HEAD 16
#define HEAD_DIM 64
#define SEQ 2048
#define BATCH 4
#define ROWS (BATCH * SEQ)      // 8192

// ---- projection GEMM tiling ----
#define BM 128
#define BN 128
#define BK 32
// 8 waves: 2 (M) x 4 (N); wave tile 64x32 -> 4x2 WMMA accumulators

// ---- attention tiling ----
#define TK 64                   // key tile width (4 x 16-col WMMA sub-tiles)

// softmax in base-2: 1/sqrt(64) * log2(e)
#define SM_SCALE 0.18033688011112042f

__device__ __forceinline__ float exp2_fast(float x) {
    return __builtin_amdgcn_exp2f(x);            // v_exp_f32 (base-2 HW exp)
}

// Build a v16h A/B fragment from two 16-byte-aligned chunks of 8 halves.
__device__ __forceinline__ v16h ld2(const _Float16* p0, const _Float16* p1) {
    v8h a = *(const v8h*)p0;
    v8h b = *(const v8h*)p1;
    v16h r;
#pragma unroll
    for (int i = 0; i < 8; ++i) { r[i] = a[i]; r[i + 8] = b[i]; }
    return r;
}

__device__ __forceinline__ v8f wmma_f16(v16h a, v16h b, v8f c) {
    return __builtin_amdgcn_wmma_f32_16x16x32_f16(false, a, false, b, (short)0, c,
                                                  false, false);
}

// ---------------------------------------------------------------------------
// Y = X @ W + bias, X:[8192,1024] f32, W:[1024,1024] f32 -> f16 output.
// VT == 0: out layout [B, H, S, D]   (Q, K)
// VT == 1: out layout [B, H, D, S]   (V, transposed for direct B-fragments)
// Double-buffered LDS pipeline: one barrier per K-step, global loads for the
// next tile overlap the WMMAs on the current tile.
// ---------------------------------------------------------------------------
template <int VT>
__global__ __launch_bounds__(256) void qkv_proj_kernel(
    const float* __restrict__ X, const float* __restrict__ W,
    const float* __restrict__ bias, _Float16* __restrict__ out)
{
    __shared__ __align__(16) _Float16 Xs[2][BM][BK + 8];   // [buf][row][k]
    __shared__ __align__(16) _Float16 Ws[2][BN][BK + 8];   // [buf][n][k] (transposed)

    const int tid  = threadIdx.x;
    const int lane = tid & 31;
    const int wave = tid >> 5;
    const int wm   = wave & 1;       // M-wave (0..1) -> 64-row strip
    const int wn   = wave >> 1;      // N-wave (0..3) -> 32-col strip
    const int bm   = blockIdx.x * BM;
    const int bn   = blockIdx.y * BN;
    const int half = lane >> 4;      // 0 | 1
    const int l16  = lane & 15;

    v8f acc[4][2] = {};
    float4 xr[4], wr[4];             // staging registers for the next tile

    auto load_tiles = [&](int k0) {
#pragma unroll
        for (int i = 0; i < 4; ++i) {
            const int idx = tid + 256 * i;          // 0..1023
            xr[i] = *(const float4*)(X + (size_t)(bm + (idx >> 3)) * N_EMBD +
                                     k0 + (idx & 7) * 4);
            wr[i] = *(const float4*)(W + (size_t)(k0 + (idx >> 5)) * N_EMBD +
                                     bn + (idx & 31) * 4);
        }
    };
    auto store_tiles = [&](int buf) {
#pragma unroll
        for (int i = 0; i < 4; ++i) {
            const int idx = tid + 256 * i;
            _Float16* dx = &Xs[buf][idx >> 3][(idx & 7) * 4];
            dx[0] = (_Float16)xr[i].x; dx[1] = (_Float16)xr[i].y;
            dx[2] = (_Float16)xr[i].z; dx[3] = (_Float16)xr[i].w;
            const int kr = idx >> 5, cg = idx & 31;
            Ws[buf][cg * 4 + 0][kr] = (_Float16)wr[i].x;
            Ws[buf][cg * 4 + 1][kr] = (_Float16)wr[i].y;
            Ws[buf][cg * 4 + 2][kr] = (_Float16)wr[i].z;
            Ws[buf][cg * 4 + 3][kr] = (_Float16)wr[i].w;
        }
    };

    load_tiles(0);
    store_tiles(0);

    int buf = 0;
    for (int k0 = 0; k0 < N_EMBD; k0 += BK) {
        __syncthreads();                          // buf is ready for all waves
        const bool more = (k0 + BK) < N_EMBD;
        if (more) load_tiles(k0 + BK);            // overlap with WMMAs below

        // A fragments: row = wm*64 + m*16 + l16; halves {ab..ab+7, ab+16..ab+23}
        const int ab = half ? 8 : 0;
        v16h a[4];
#pragma unroll
        for (int m = 0; m < 4; ++m) {
            const _Float16* rp = &Xs[buf][wm * 64 + m * 16 + l16][0];
            a[m] = ld2(rp + ab, rp + ab + 16);
        }
        // B fragments: col = wn*32 + n*16 + l16; halves {bb..bb+7, bb+16..bb+23}
        const int bb = half ? 8 : 0;
        v16h b[2];
#pragma unroll
        for (int n = 0; n < 2; ++n) {
            const _Float16* rp = &Ws[buf][wn * 32 + n * 16 + l16][0];
            b[n] = ld2(rp + bb, rp + bb + 16);
        }
#pragma unroll
        for (int m = 0; m < 4; ++m)
#pragma unroll
            for (int n = 0; n < 2; ++n)
                acc[m][n] = wmma_f16(a[m], b[n], acc[m][n]);

        if (more) store_tiles(buf ^ 1);           // prior reads of buf^1 ended
        buf ^= 1;                                 // before this iter's barrier
    }

    // epilogue: +bias, store f16
#pragma unroll
    for (int n = 0; n < 2; ++n) {
        const int col = bn + wn * 32 + n * 16 + l16;         // 0..1023
        const float bv = bias[col];
        const int h = col >> 6;
        const int d = col & 63;
#pragma unroll
        for (int m = 0; m < 4; ++m) {
            if (VT) {
                // [B, H, D, S]: 8 accumulator rows = 8 consecutive seq -> v8h
                const int row0 = bm + wm * 64 + m * 16 + 8 * half;
                const int bidx = row0 >> 11;
                const int s0   = row0 & 2047;
                v8h pk;
#pragma unroll
                for (int r = 0; r < 8; ++r) pk[r] = (_Float16)(acc[m][n][r] + bv);
                *(v8h*)(out + ((size_t)((bidx * N_HEAD + h) * HEAD_DIM + d)) * SEQ + s0) = pk;
            } else {
                // [B, H, S, D]
#pragma unroll
                for (int r = 0; r < 8; ++r) {
                    const int row  = bm + wm * 64 + m * 16 + r + 8 * half;
                    const int bidx = row >> 11;
                    const int s    = row & 2047;
                    out[(((size_t)(bidx * N_HEAD + h)) * SEQ + s) * HEAD_DIM + d] =
                        (_Float16)(acc[m][n][r] + bv);
                }
            }
        }
    }
}

// ---------------------------------------------------------------------------
// Flash attention: per (b,h) head, 128 query rows per block (16 per wave).
// Q,K in [B,H,S,D] f16; V in [B,H,D,S] f16 (transposed). All operand
// fragments come straight from global memory -> no barriers in the K-loop;
// only a per-wave LDS bounce converts P from C-layout to A-fragment layout.
// TK=64 amortizes the softmax reduction trees and o-rescale over 2x columns.
// ---------------------------------------------------------------------------
__global__ __launch_bounds__(256) void attn_kernel(
    const _Float16* __restrict__ Qh, const _Float16* __restrict__ Kh,
    const _Float16* __restrict__ Vt, float* __restrict__ out)
{
    __shared__ __align__(16) _Float16 Ps[8][16][TK];        // per-wave P scratch

    const int tid  = threadIdx.x;
    const int lane = tid & 31;
    const int wave = tid >> 5;
    const int half = lane >> 4;
    const int l16  = lane & 15;
    const int bh   = blockIdx.y;                 // b*16 + h
    const int q0   = blockIdx.x * 128 + wave * 16;

    const _Float16* Qp  = Qh + (size_t)bh * SEQ * HEAD_DIM;
    const _Float16* Kp  = Kh + (size_t)bh * SEQ * HEAD_DIM;
    const _Float16* Vtp = Vt + (size_t)bh * HEAD_DIM * SEQ;

    // Q A-fragments (head dim split into two K=32 chunks), loaded once
    v16h aq[2];
    {
        const int ab = half ? 8 : 0;
        const _Float16* rq = Qp + (size_t)(q0 + l16) * HEAD_DIM;
        aq[0] = ld2(rq + ab,      rq + ab + 16);
        aq[1] = ld2(rq + 32 + ab, rq + 32 + ab + 16);
    }

    v8f o[4] = {};
    float mrow[8], lrow[8];
#pragma unroll
    for (int r = 0; r < 8; ++r) { mrow[r] = -3.0e38f; lrow[r] = 0.0f; }

    const int bb = half ? 8 : 0;     // K: head-dims {bb..bb+7, bb+16..bb+23} per lane
    const int vb = half ? 8 : 0;     // V: seq positions {vb..vb+7, vb+16..vb+23} per lane
    const int ab = half ? 8 : 0;

    for (int kt = 0; kt < SEQ; kt += TK) {
        // ---- scores S = Q K^T for a 16x64 tile (four 16-col sub-tiles) ----
        v8f sc[4] = {};
        {
            const _Float16* kr[4];
#pragma unroll
            for (int c = 0; c < 4; ++c) {
                kr[c] = Kp + (size_t)(kt + c * 16 + l16) * HEAD_DIM + bb;
                __builtin_prefetch(kr[c] + (size_t)TK * HEAD_DIM, 0, 3);
            }
#pragma unroll
            for (int kk = 0; kk < 2; ++kk)
#pragma unroll
                for (int c = 0; c < 4; ++c) {
                    v16h b = ld2(kr[c] + kk * 32, kr[c] + kk * 32 + 16);
                    sc[c] = wmma_f16(aq[kk], b, sc[c]);
                }
        }

        // ---- online softmax (base-2): row stats across 16-lane half-wave ----
#pragma unroll
        for (int r = 0; r < 8; ++r) {
            float s[4];
#pragma unroll
            for (int c = 0; c < 4; ++c) s[c] = sc[c][r] * SM_SCALE;
            float t = fmaxf(fmaxf(s[0], s[1]), fmaxf(s[2], s[3]));
#pragma unroll
            for (int msk = 1; msk < 16; msk <<= 1)
                t = fmaxf(t, __shfl_xor(t, msk, 32));
            const float mnew  = fmaxf(mrow[r], t);
            const float alpha = exp2_fast(mrow[r] - mnew);
            float p[4], rs = 0.0f;
#pragma unroll
            for (int c = 0; c < 4; ++c) { p[c] = exp2_fast(s[c] - mnew); rs += p[c]; }
#pragma unroll
            for (int msk = 1; msk < 16; msk <<= 1)
                rs += __shfl_xor(rs, msk, 32);
            lrow[r] = lrow[r] * alpha + rs;
            mrow[r] = mnew;
#pragma unroll
            for (int t4 = 0; t4 < 4; ++t4) o[t4][r] *= alpha;
            // spill P (C-layout) to per-wave LDS for A-fragment re-read
#pragma unroll
            for (int c = 0; c < 4; ++c)
                Ps[wave][r + 8 * half][c * 16 + l16] = (_Float16)p[c];
        }

        // ---- P @ V: A = P (16x64, two K=32 chunks), B direct from V^T ----
#pragma unroll
        for (int ka = 0; ka < 2; ++ka) {
            const _Float16* pr = &Ps[wave][l16][ka * 32];
            v16h ap = ld2(pr + ab, pr + ab + 16);
#pragma unroll
            for (int t4 = 0; t4 < 4; ++t4) {
                const _Float16* vr =
                    Vtp + (size_t)(t4 * 16 + l16) * SEQ + kt + ka * 32 + vb;
                if (ka == 0) __builtin_prefetch(vr + TK, 0, 3);
                v16h bv = ld2(vr, vr + 16);
                o[t4] = wmma_f16(ap, bv, o[t4]);
            }
        }
    }

    // ---- normalize and write fp32 output [B, S, E] ----
    const int b = bh >> 4;
    const int h = bh & 15;
#pragma unroll
    for (int t4 = 0; t4 < 4; ++t4) {
        const int col = h * HEAD_DIM + t4 * 16 + l16;
#pragma unroll
        for (int r = 0; r < 8; ++r) {
            const int s = q0 + r + 8 * half;
            out[((size_t)b * SEQ + s) * N_EMBD + col] = o[t4][r] / lrow[r];
        }
    }
}

// ---------------------------------------------------------------------------
extern "C" void kernel_launch(void* const* d_in, const int* in_sizes, int n_in,
                              void* d_out, int out_size, void* d_ws, size_t ws_size,
                              hipStream_t stream) {
    (void)in_sizes; (void)n_in; (void)out_size; (void)ws_size;
    const float* X  = (const float*)d_in[0];
    const float* Wq = (const float*)d_in[1];
    const float* bq = (const float*)d_in[2];
    const float* Wk = (const float*)d_in[3];
    const float* bk = (const float*)d_in[4];
    const float* Wv = (const float*)d_in[5];
    const float* bv = (const float*)d_in[6];
    float* out = (float*)d_out;

    const size_t qkv_elems = (size_t)BATCH * N_HEAD * SEQ * HEAD_DIM; // 8,388,608
    _Float16* Qh  = (_Float16*)d_ws;
    _Float16* Kh  = Qh + qkv_elems;
    _Float16* Vth = Kh + qkv_elems;

    dim3 gp(ROWS / BM, N_EMBD / BN);      // 64 x 8 blocks
    qkv_proj_kernel<0><<<gp, 256, 0, stream>>>(X, Wq, bq, Qh);
    qkv_proj_kernel<0><<<gp, 256, 0, stream>>>(X, Wk, bk, Kh);
    qkv_proj_kernel<1><<<gp, 256, 0, stream>>>(X, Wv, bv, Vth);

    dim3 ga(SEQ / 128, BATCH * N_HEAD);   // 16 x 64 blocks
    attn_kernel<<<ga, 256, 0, stream>>>(Qh, Kh, Vth, out);
}
